// TimesFMAttention_48473000903586
// MI455X (gfx1250) — compile-verified, matrix operand order checked by reading, not run on hardware
//
#include <hip/hip_runtime.h>
#include <hip/hip_bf16.h>

#define NUM_HEADS 16
#define NUM_KV    4
#define HEAD_DIM  64
#define HIDDEN    1024
#define SEQ       2048
#define BATCH     2
#define QKV_OUT   ((NUM_HEADS + 2 * NUM_KV) * HEAD_DIM)   // 1536
#define MTOT      (BATCH * SEQ)                            // 4096

typedef __bf16  v16bf  __attribute__((ext_vector_type(16)));
typedef float   v8f    __attribute__((ext_vector_type(8)));
typedef unsigned int uint4v __attribute__((ext_vector_type(4)));

union Frag {
    uint4v u4[2];
    v16bf  v;
};

__device__ __forceinline__ unsigned short f2bf(float f) {
    unsigned int u = __float_as_uint(f);
    unsigned int r = u + 0x7FFFu + ((u >> 16) & 1u);   // round-to-nearest-even
    return (unsigned short)(r >> 16);
}

// Load one 16-element bf16 fragment: two contiguous 16B chunks 32B apart.
// Works for both global and LDS-derived pointers (16B aligned).
__device__ __forceinline__ void load_frag(const unsigned short* p, Frag& f) {
    const uint4v* q = reinterpret_cast<const uint4v*>(p);
    f.u4[0] = q[0];
    f.u4[1] = q[2];
}

__device__ __forceinline__ v8f wmma_bf16(const Frag& a, const Frag& b, v8f c) {
    return __builtin_amdgcn_wmma_f32_16x16x32_bf16(
        false, a.v, false, b.v, (short)0, c, false, false);
}

// ---------------------------------------------------------------- conversion
__global__ void cvt_bf16_kernel(const float* __restrict__ src,
                                unsigned short* __restrict__ dst, int n) {
    int i = blockIdx.x * blockDim.x + threadIdx.x;
    if (i < n) dst[i] = f2bf(src[i]);
}

// ---------------------------------------------------------------- QKV GEMM
// A: [4096,1024] bf16 row-major.  W: [1536,1024] bf16 row-major (C = A W^T + b)
// Block tile 64x256, 8 waves (2x4), wave tile 32x64 (2x4 WMMA accumulators).
// Epilogue scatters Q (scaled), K, and V^T into attention layouts.
__global__ __launch_bounds__(256)
void qkv_gemm_kernel(const unsigned short* __restrict__ A,
                     const unsigned short* __restrict__ W,
                     const float* __restrict__ bias,
                     const float* __restrict__ scaling,
                     unsigned short* __restrict__ q_bf,   // [B,H,S,64]
                     unsigned short* __restrict__ k_bf,   // [B,KV,S,64]
                     unsigned short* __restrict__ vt_bf)  // [B,KV,64,S]
{
    const int K    = HIDDEN;
    const int wave = threadIdx.x >> 5;
    const int lane = threadIdx.x & 31;
    const int wm = wave >> 2, wn = wave & 3;
    const int m0 = blockIdx.x * 64  + wm * 32;
    const int n0 = blockIdx.y * 256 + wn * 64;
    const int r = lane & 15, g = lane >> 4, kofs = g * 8;

    v8f acc[2][4] = {};
    for (int k0 = 0; k0 < K; k0 += 32) {
        Frag a[2], b[4];
#pragma unroll
        for (int i = 0; i < 2; ++i)
            load_frag(A + (size_t)(m0 + i * 16 + r) * K + k0 + kofs, a[i]);
#pragma unroll
        for (int j = 0; j < 4; ++j)
            load_frag(W + (size_t)(n0 + j * 16 + r) * K + k0 + kofs, b[j]);
#pragma unroll
        for (int i = 0; i < 2; ++i)
#pragma unroll
            for (int j = 0; j < 4; ++j)
                acc[i][j] = wmma_bf16(a[i], b[j], acc[i][j]);
    }

#pragma unroll
    for (int j = 0; j < 4; ++j) {
        const int n  = n0 + j * 16 + r;
        const float bn = bias[n];
        if (n < NUM_HEADS * HEAD_DIM) {                      // Q + softplus scale
            const int h = n >> 6, d = n & 63;
            const float sv = scaling[d];
            const float sp = (sv > 20.0f) ? sv : log1pf(expf(sv));
            const float qs = (1.442695041f / 8.0f) * sp;     // R/sqrt(64)
#pragma unroll
            for (int i = 0; i < 2; ++i)
#pragma unroll
                for (int v = 0; v < 8; ++v) {
                    const int m = m0 + i * 16 + g * 8 + v;
                    const int bb = m >> 11, s = m & (SEQ - 1);
                    q_bf[(((size_t)(bb * NUM_HEADS + h)) * SEQ + s) * HEAD_DIM + d] =
                        f2bf((acc[i][j][v] + bn) * qs);
                }
        } else if (n < (NUM_HEADS + NUM_KV) * HEAD_DIM) {    // K
            const int nn = n - NUM_HEADS * HEAD_DIM;
            const int kvh = nn >> 6, d = nn & 63;
#pragma unroll
            for (int i = 0; i < 2; ++i)
#pragma unroll
                for (int v = 0; v < 8; ++v) {
                    const int m = m0 + i * 16 + g * 8 + v;
                    const int bb = m >> 11, s = m & (SEQ - 1);
                    k_bf[(((size_t)(bb * NUM_KV + kvh)) * SEQ + s) * HEAD_DIM + d] =
                        f2bf(acc[i][j][v] + bn);
                }
        } else {                                             // V, stored transposed
            const int nn = n - (NUM_HEADS + NUM_KV) * HEAD_DIM;
            const int kvh = nn >> 6, d = nn & 63;
#pragma unroll
            for (int i = 0; i < 2; ++i)
#pragma unroll
                for (int v = 0; v < 8; ++v) {
                    const int m = m0 + i * 16 + g * 8 + v;
                    const int bb = m >> 11, s = m & (SEQ - 1);
                    vt_bf[(((size_t)(bb * NUM_KV + kvh)) * HEAD_DIM + d) * SEQ + s] =
                        f2bf(acc[i][j][v] + bn);
                }
        }
    }
}

// ---------------------------------------------------------------- attention
// One wave per 16-query tile; flash-style online softmax over S in 32-key steps.
__global__ __launch_bounds__(256)
void attn_kernel(const unsigned short* __restrict__ q_bf,   // [B,H,S,64]
                 const unsigned short* __restrict__ k_bf,   // [B,KV,S,64]
                 const unsigned short* __restrict__ vt_bf,  // [B,KV,64,S]
                 unsigned short* __restrict__ attn_bf)      // [B*S, 1024]
{
    __shared__ __align__(16) unsigned short sm[8 * 16 * 32]; // per-wave P tile

    const int wave = threadIdx.x >> 5;
    const int lane = threadIdx.x & 31;
    const int tile = blockIdx.x * 8 + wave;

    const int qpb = SEQ / 16;                     // 128 q-tiles per (b,h)
    const int b   = tile / (NUM_HEADS * qpb);
    const int rem = tile % (NUM_HEADS * qpb);
    const int h   = rem / qpb;
    const int qt  = rem % qpb;
    const int kvh = h >> 2;                       // H/KV = 4

    const int r = lane & 15, g = lane >> 4, kofs = g * 8;
    unsigned short* smw = sm + wave * 512;

    // Q fragments (16 x 64 = two K=32 chunks), loaded once
    const unsigned short* qptr =
        q_bf + (((size_t)(b * NUM_HEADS + h)) * SEQ + qt * 16 + r) * HEAD_DIM;
    Frag qf[2];
    load_frag(qptr + 0 * 32 + kofs, qf[0]);
    load_frag(qptr + 1 * 32 + kofs, qf[1]);

    const unsigned short* kbase = k_bf + ((size_t)(b * NUM_KV + kvh)) * SEQ * HEAD_DIM;
    const unsigned short* vbase = vt_bf + ((size_t)(b * NUM_KV + kvh)) * HEAD_DIM * SEQ;

    v8f  o[4] = {};
    float mrow[8], lrow[8];
#pragma unroll
    for (int v = 0; v < 8; ++v) { mrow[v] = -1e30f; lrow[v] = 0.0f; }

    for (int kb0 = 0; kb0 < SEQ; kb0 += 32) {
        // scores for 32 keys: two 16x16 tiles, each K=64 via chained WMMA
        v8f s[2] = {};
#pragma unroll
        for (int t = 0; t < 2; ++t) {
            const unsigned short* kp = kbase + (size_t)(kb0 + t * 16 + r) * HEAD_DIM;
            Frag kf0, kf1;
            load_frag(kp + kofs, kf0);
            load_frag(kp + 32 + kofs, kf1);
            s[t] = wmma_bf16(qf[0], kf0, s[t]);
            s[t] = wmma_bf16(qf[1], kf1, s[t]);
        }

        // online softmax update (row = g*8+v; 16 lanes of a half hold one row)
#pragma unroll
        for (int v = 0; v < 8; ++v) {
            float x = fmaxf(s[0][v], s[1][v]);
            x = fmaxf(x, __shfl_xor(x, 1, 32));
            x = fmaxf(x, __shfl_xor(x, 2, 32));
            x = fmaxf(x, __shfl_xor(x, 4, 32));
            x = fmaxf(x, __shfl_xor(x, 8, 32));
            const float mn    = fmaxf(mrow[v], x);
            const float alpha = __expf(mrow[v] - mn);
            const float p0 = __expf(s[0][v] - mn);
            const float p1 = __expf(s[1][v] - mn);
            float rs = p0 + p1;
            rs += __shfl_xor(rs, 1, 32);
            rs += __shfl_xor(rs, 2, 32);
            rs += __shfl_xor(rs, 4, 32);
            rs += __shfl_xor(rs, 8, 32);
            lrow[v] = lrow[v] * alpha + rs;
            mrow[v] = mn;
#pragma unroll
            for (int dt = 0; dt < 4; ++dt) o[dt][v] *= alpha;
            // stash P (bf16) into LDS in [row][key] order for re-layout
            smw[(g * 8 + v) * 32 + r]      = f2bf(p0);
            smw[(g * 8 + v) * 32 + 16 + r] = f2bf(p1);
        }

        asm volatile("s_wait_dscnt 0" ::: "memory");   // DS stores visible to reads

        // P as A-fragment (16x32 over the 32 keys)
        Frag pf;
        load_frag(smw + r * 32 + kofs, pf);

        // out += P x V  (Vt rows are contiguous in key index -> b128 loads)
#pragma unroll
        for (int dt = 0; dt < 4; ++dt) {
            Frag vf;
            load_frag(vbase + (size_t)(dt * 16 + r) * SEQ + kb0 + kofs, vf);
            o[dt] = wmma_bf16(pf, vf, o[dt]);
        }
    }

    // normalize + store [B*S, H*64] bf16
#pragma unroll
    for (int dt = 0; dt < 4; ++dt)
#pragma unroll
        for (int v = 0; v < 8; ++v) {
            const int m = qt * 16 + g * 8 + v;
            attn_bf[((size_t)(b * SEQ + m)) * (NUM_HEADS * HEAD_DIM)
                    + h * HEAD_DIM + dt * 16 + r] = f2bf(o[dt][v] / lrow[v]);
        }
}

// ---------------------------------------------------------------- O GEMM
// Block tile 64x256, wave tile 32x64, fp32 output + bias.
__global__ __launch_bounds__(256)
void o_gemm_kernel(const unsigned short* __restrict__ A,   // [4096,1024] bf16
                   const unsigned short* __restrict__ W,   // [1024,1024] bf16
                   const float* __restrict__ bias,         // [1024]
                   float* __restrict__ out)                // [4096,1024] f32
{
    const int K    = NUM_HEADS * HEAD_DIM;   // 1024
    const int wave = threadIdx.x >> 5;
    const int lane = threadIdx.x & 31;
    const int wm = wave >> 2, wn = wave & 3;
    const int m0 = blockIdx.x * 64  + wm * 32;
    const int n0 = blockIdx.y * 256 + wn * 64;
    const int r = lane & 15, g = lane >> 4, kofs = g * 8;

    v8f acc[2][4] = {};
    for (int k0 = 0; k0 < K; k0 += 32) {
        Frag a[2], b[4];
#pragma unroll
        for (int i = 0; i < 2; ++i)
            load_frag(A + (size_t)(m0 + i * 16 + r) * K + k0 + kofs, a[i]);
#pragma unroll
        for (int j = 0; j < 4; ++j)
            load_frag(W + (size_t)(n0 + j * 16 + r) * K + k0 + kofs, b[j]);
#pragma unroll
        for (int i = 0; i < 2; ++i)
#pragma unroll
            for (int j = 0; j < 4; ++j)
                acc[i][j] = wmma_bf16(a[i], b[j], acc[i][j]);
    }

#pragma unroll
    for (int j = 0; j < 4; ++j) {
        const int n = n0 + j * 16 + r;
        const float bn = bias[n];
#pragma unroll
        for (int i = 0; i < 2; ++i)
#pragma unroll
            for (int v = 0; v < 8; ++v) {
                const int m = m0 + i * 16 + g * 8 + v;
                out[(size_t)m * HIDDEN + n] = acc[i][j][v] + bn;
            }
    }
}

// ---------------------------------------------------------------- launch
extern "C" void kernel_launch(void* const* d_in, const int* in_sizes, int n_in,
                              void* d_out, int out_size, void* d_ws, size_t ws_size,
                              hipStream_t stream) {
    const float* hidden  = (const float*)d_in[0];
    const float* scaling = (const float*)d_in[1];
    const float* qkv_w   = (const float*)d_in[2];
    const float* qkv_b   = (const float*)d_in[3];
    const float* o_w     = (const float*)d_in[4];
    const float* o_b     = (const float*)d_in[5];
    float* out = (float*)d_out;

    char* ws = (char*)d_ws;
    size_t off = 0;
    unsigned short* hid_bf  = (unsigned short*)(ws + off); off += (size_t)MTOT * HIDDEN * 2;
    unsigned short* qkvw_bf = (unsigned short*)(ws + off); off += (size_t)QKV_OUT * HIDDEN * 2;
    unsigned short* ow_bf   = (unsigned short*)(ws + off); off += (size_t)HIDDEN * HIDDEN * 2;
    unsigned short* q_bf    = (unsigned short*)(ws + off); off += (size_t)BATCH * NUM_HEADS * SEQ * HEAD_DIM * 2;
    unsigned short* k_bf    = (unsigned short*)(ws + off); off += (size_t)BATCH * NUM_KV * SEQ * HEAD_DIM * 2;
    unsigned short* vt_bf   = (unsigned short*)(ws + off); off += (size_t)BATCH * NUM_KV * HEAD_DIM * SEQ * 2;
    unsigned short* attn_bf = (unsigned short*)(ws + off); off += (size_t)MTOT * NUM_HEADS * HEAD_DIM * 2;
    (void)in_sizes; (void)n_in; (void)out_size; (void)ws_size;

    {
        int n = MTOT * HIDDEN;
        cvt_bf16_kernel<<<(n + 255) / 256, 256, 0, stream>>>(hidden, hid_bf, n);
    }
    {
        int n = QKV_OUT * HIDDEN;
        cvt_bf16_kernel<<<(n + 255) / 256, 256, 0, stream>>>(qkv_w, qkvw_bf, n);
    }
    {
        int n = HIDDEN * HIDDEN;
        cvt_bf16_kernel<<<(n + 255) / 256, 256, 0, stream>>>(o_w, ow_bf, n);
    }

    qkv_gemm_kernel<<<dim3(MTOT / 64, QKV_OUT / 256), 256, 0, stream>>>(
        hid_bf, qkvw_bf, qkv_b, scaling, q_bf, k_bf, vt_bf);

    attn_kernel<<<(BATCH * NUM_HEADS * (SEQ / 16)) / 8, 256, 0, stream>>>(
        q_bf, k_bf, vt_bf, attn_bf);

    o_gemm_kernel<<<dim3(MTOT / 64, HIDDEN / 256), 256, 0, stream>>>(
        attn_bf, ow_bf, o_b, out);
}
